// WTM_23587960390302
// MI455X (gfx1250) — matrix operand order repaired and checked by reading, not compiled, not run on hardware
//
#include <hip/hip_runtime.h>
#include <hip/hip_bf16.h>
#include <math.h>

// ---------- types ----------
typedef __bf16 bf16_t;
typedef __attribute__((ext_vector_type(16))) __bf16 v16bf;
typedef __attribute__((ext_vector_type(8)))  float  v8f;

// ---------- problem constants ----------
#define BB   2
#define CC   64
#define HH   384
#define WW   384
#define HWP  (HH * WW)        // 147456
#define WSZ  8
#define SS   64               // tokens per window
#define NHD  4
#define HDD  16
#define NWX  48
#define NWY  48

// ---------- padded LDS strides (64 banks x 4B; avoid stride%64==0 in elements) ----------
#define A_LDX 68   // f32 raw window tile
#define A_LDQ 200  // f32 qkv tile
#define A_LDR 68   // f32 r / xc tiles
#define A_LDB 72   // bf16 A tiles (row stride 144B, 16B aligned)
#define B1_LDB 72
#define B3_LDY 264 // bf16 y2 tile (row stride 528B, 16B aligned, banks coprime-ish)

// ---------- optional gfx1250 async global->LDS path ----------
#if defined(__has_builtin)
#if __has_builtin(__builtin_amdgcn_global_load_async_to_lds_b128) && \
    __has_builtin(__builtin_amdgcn_s_wait_asynccnt)
#define HAVE_ASYNC_LDS 1
#endif
#endif

#ifdef HAVE_ASYNC_LDS
typedef int v4i_t __attribute__((vector_size(4 * sizeof(int))));
typedef __attribute__((address_space(1))) v4i_t* gv4i_p;   // global int4*
typedef __attribute__((address_space(3))) v4i_t* lv4i_p;   // LDS int4*
__device__ __forceinline__ void async_ld_b128(const void* g, void* l) {
  __builtin_amdgcn_global_load_async_to_lds_b128(
      (gv4i_p)(unsigned long long)g,
      (lv4i_p)(unsigned)(unsigned long long)l, 0, 0);
}
__device__ __forceinline__ void async_wait0() { __builtin_amdgcn_s_wait_asynccnt(0); }
#endif

__device__ __forceinline__ int laneid() { return threadIdx.x & 31; }
__device__ __forceinline__ int waveid() {
  return __builtin_amdgcn_readfirstlane((int)threadIdx.x) >> 5;  // scalar (SGPR)
}

// A fragment (16x32 bf16) from row-major LDS [..][ld], tile origin (row0, k0).
// ISA: lanes 0-15 -> row=lane, K {0..7,16..23}; lanes 16-31 -> row=lane-16, K {8..15,24..31}
__device__ __forceinline__ v16bf ldA(const bf16_t* A, int ld, int row0, int k0) {
  int l = laneid();
  const bf16_t* p = A + (size_t)(row0 + (l & 15)) * ld + k0 + ((l >> 4) << 3);
  v16bf a;
#pragma unroll
  for (int i = 0; i < 8; ++i) a[i] = p[i];
#pragma unroll
  for (int i = 0; i < 8; ++i) a[i + 8] = p[i + 16];
  return a;
}

// B fragment (32x16 bf16) with B[k][n] = W[n][k], W row-major [N][ldk] (global).
// ISA: lanes 0-15 -> col=lane, K {0..7,16..23}; lanes 16-31 -> col=lane-16, K {8..15,24..31}
__device__ __forceinline__ v16bf ldBT(const bf16_t* Wt, int ldk, int n0, int k0) {
  int l = laneid();
  const bf16_t* p = Wt + (size_t)(n0 + (l & 15)) * ldk + k0 + ((l >> 4) << 3);
  v16bf b;
#pragma unroll
  for (int i = 0; i < 16; ++i) b[i] = p[i + ((i >> 3) << 3)];
  return b;
}

// one 16x16 D tile of A(MxK) * W^T(KxN), K multiple of 32, f32 accumulate
__device__ __forceinline__ v8f wmma_tile(const bf16_t* A, int lda, int row0,
                                         const bf16_t* Wt, int ldk, int n0, int Ktot) {
  v8f acc = {};
  for (int k0 = 0; k0 < Ktot; k0 += 32) {
    v16bf a = ldA(A, lda, row0, k0);
    v16bf b = ldBT(Wt, ldk, n0, k0);
    acc = __builtin_amdgcn_wmma_f32_16x16x32_bf16(false, a, false, b, (short)0, acc,
                                                  false, false);
  }
  return acc;
}

// ---------- f32 -> bf16 weight conversion ----------
__global__ void cvt_bf16(const float* __restrict__ s, bf16_t* __restrict__ d, int n) {
  int i = blockIdx.x * 256 + threadIdx.x;
  if (i < n) d[i] = (bf16_t)s[i];
}

// ---------- Stage A : LN1 + windowed ReMSA (+conv branch) + residual ----------
__global__ __launch_bounds__(256) void stageA(
    const float* __restrict__ x,
    const float* __restrict__ n1w, const float* __restrict__ n1b,
    const bf16_t* __restrict__ wq,  const float* __restrict__ bq,
    const float* __restrict__ posb,
    const bf16_t* __restrict__ wo,  const float* __restrict__ bo,
    const bf16_t* __restrict__ wcp, const float* __restrict__ bcp,
    const bf16_t* __restrict__ wp1, const float* __restrict__ bp1,
    const float* __restrict__ dw1,  const float* __restrict__ dw1b,
    const bf16_t* __restrict__ wtk, const float* __restrict__ btk,
    float* __restrict__ x1)
{
  extern __shared__ char smem[];
  float*  s_x   = (float*)smem;                 // 64 x A_LDX f32
  float*  s_qkv = s_x + SS * A_LDX;             // 64 x A_LDQ f32
  float*  s_r   = s_qkv + SS * A_LDQ;           // 64 x A_LDR f32
  float*  s_t1  = s_r + SS * A_LDR;
  float*  s_t2  = s_t1 + SS * A_LDR;
  bf16_t* s_xn  = (bf16_t*)(s_t2 + SS * A_LDR); // 64 x A_LDB bf16
  bf16_t* s_ab  = s_xn + SS * A_LDB;
  bf16_t* s_tb  = s_ab + SS * A_LDB;

  const int tid = threadIdx.x;
  const int wv  = waveid();
  const int blk = blockIdx.x;
  const int b   = blk / (NWX * NWY);
  const int wi  = blk % (NWX * NWY);
  const int wy  = wi / NWX, wx = wi % NWX;
  const float* xb = x + (size_t)b * CC * HWP;
  const int base_hw = (wy * WSZ) * WW + wx * WSZ;

  // load window: token-major (p, ch)
#pragma unroll
  for (int i = 0; i < 16; ++i) {
    int idx = i * 256 + tid;
    int ch = idx >> 6, p = idx & 63;
    int r = p >> 3, c = p & 7;
    s_x[p * A_LDX + ch] = xb[(size_t)ch * HWP + base_hw + r * WW + c];
  }
  __syncthreads();

  // channel LayerNorm per token (biased variance)
  if (tid < SS) {
    const float* row = s_x + tid * A_LDX;
    float mu = 0.f;
    for (int ch = 0; ch < CC; ++ch) mu += row[ch];
    mu *= (1.f / CC);
    float var = 0.f;
    for (int ch = 0; ch < CC; ++ch) { float d = row[ch] - mu; var += d * d; }
    var *= (1.f / CC);
    float inv = rsqrtf(var + 1e-5f);
    for (int ch = 0; ch < CC; ++ch)
      s_xn[tid * A_LDB + ch] = (bf16_t)((row[ch] - mu) * inv * n1w[ch] + n1b[ch]);
  }
  __syncthreads();

  // qkv GEMM: (64x64) @ (64x192) -> 4x12 tiles
  for (int t = wv; t < 48; t += 8) {
    int rt = t & 3, ct = t >> 2;
    v8f acc = wmma_tile(s_xn, A_LDB, rt * 16, wq, CC, ct * 16, CC);
    int l = laneid();
    int col = ct * 16 + (l & 15);
    int r0  = rt * 16 + ((l >> 4) << 3);
    float bv = bq[col];
#pragma unroll
    for (int i = 0; i < 8; ++i) s_qkv[(r0 + i) * A_LDQ + col] = acc[i] + bv;
  }
  __syncthreads();

  // attention: per (token, head); heads-mixing 4x4 scores (faithful to reference)
  {
    int t = tid >> 2, h = tid & 3;
    const float* qv = s_qkv + t * A_LDQ;   // [head][q16 k16 v16]
    float sc[4];
    float mx = -1e30f;
#pragma unroll
    for (int g = 0; g < 4; ++g) {
      float s = 0.f;
#pragma unroll
      for (int d = 0; d < HDD; ++d) s += qv[h * 48 + d] * qv[g * 48 + 16 + d];
      s = s * 0.25f + posb[t * 16 + h * 4 + g];   // HD^-0.5 = 0.25
      sc[g] = s; mx = fmaxf(mx, s);
    }
    float den = 0.f;
#pragma unroll
    for (int g = 0; g < 4; ++g) { sc[g] = __expf(sc[g] - mx); den += sc[g]; }
    float dinv = 1.f / den;
#pragma unroll
    for (int d = 0; d < HDD; ++d) {
      float o = 0.f;
#pragma unroll
      for (int g = 0; g < 4; ++g) o += sc[g] * qv[g * 48 + 32 + d];
      s_ab[t * A_LDB + h * HDD + d] = (bf16_t)(o * dinv);
    }
  }
  __syncthreads();

  // out-proj GEMM -> s_r  ;  convp GEMM -> s_t1
  for (int t = wv; t < 16; t += 8) {
    int rt = t & 3, ct = t >> 2;
    v8f acc = wmma_tile(s_ab, A_LDB, rt * 16, wo, CC, ct * 16, CC);
    int l = laneid(); int col = ct * 16 + (l & 15); int r0 = rt * 16 + ((l >> 4) << 3);
    float bv = bo[col];
#pragma unroll
    for (int i = 0; i < 8; ++i) s_r[(r0 + i) * A_LDR + col] = acc[i] + bv;
  }
  for (int t = wv; t < 16; t += 8) {
    int rt = t & 3, ct = t >> 2;
    v8f acc = wmma_tile(s_xn, A_LDB, rt * 16, wcp, CC, ct * 16, CC);
    int l = laneid(); int col = ct * 16 + (l & 15); int r0 = rt * 16 + ((l >> 4) << 3);
    float bv = bcp[col];
#pragma unroll
    for (int i = 0; i < 8; ++i) s_t1[(r0 + i) * A_LDR + col] = acc[i] + bv;
  }
  __syncthreads();

  // xc1 -> bf16
#pragma unroll
  for (int i = 0; i < 16; ++i) {
    int idx = i * 256 + tid;
    int s = idx >> 6, ch = idx & 63;
    s_tb[s * A_LDB + ch] = (bf16_t)s_t1[s * A_LDR + ch];
  }
  __syncthreads();

  // pw1 1x1 GEMM -> s_t2
  for (int t = wv; t < 16; t += 8) {
    int rt = t & 3, ct = t >> 2;
    v8f acc = wmma_tile(s_tb, A_LDB, rt * 16, wp1, CC, ct * 16, CC);
    int l = laneid(); int col = ct * 16 + (l & 15); int r0 = rt * 16 + ((l >> 4) << 3);
    float bv = bp1[col];
#pragma unroll
    for (int i = 0; i < 8; ++i) s_t2[(r0 + i) * A_LDR + col] = acc[i] + bv;
  }
  __syncthreads();

  // depthwise conv1d k=3 along token axis (SAME, zero pad), write bf16
#pragma unroll
  for (int i = 0; i < 16; ++i) {
    int idx = i * 256 + tid;
    int s = idx >> 6, ch = idx & 63;
    float a0 = (s > 0)      ? s_t2[(s - 1) * A_LDR + ch] : 0.f;
    float a1 =                s_t2[s * A_LDR + ch];
    float a2 = (s < SS - 1) ? s_t2[(s + 1) * A_LDR + ch] : 0.f;
    float r = a0 * dw1[ch * 3 + 0] + a1 * dw1[ch * 3 + 1] + a2 * dw1[ch * 3 + 2] + dw1b[ch];
    s_tb[s * A_LDB + ch] = (bf16_t)r;
  }
  __syncthreads();

  // tok GEMM, accumulate into s_r
  for (int t = wv; t < 16; t += 8) {
    int rt = t & 3, ct = t >> 2;
    v8f acc = wmma_tile(s_tb, A_LDB, rt * 16, wtk, CC, ct * 16, CC);
    int l = laneid(); int col = ct * 16 + (l & 15); int r0 = rt * 16 + ((l >> 4) << 3);
    float bv = btk[col];
#pragma unroll
    for (int i = 0; i < 8; ++i) s_r[(r0 + i) * A_LDR + col] += acc[i] + bv;
  }
  __syncthreads();

  // residual: x1 = x + r  (window merge implicit: windows tile the image)
  float* ob = x1 + (size_t)b * CC * HWP;
#pragma unroll
  for (int i = 0; i < 16; ++i) {
    int idx = i * 256 + tid;
    int ch = idx >> 6, p = idx & 63;
    int r = p >> 3, c = p & 7;
    ob[(size_t)ch * HWP + base_hw + r * WW + c] = s_x[p * A_LDX + ch] + s_r[p * A_LDR + ch];
  }
}

// ---------- Stage B1 : LN2 + LeFF 1x1 conv (no bias) ----------
// s_x kept CHANNEL-major [64][256]: global<->LDS linear (async-friendly),
// LN reads are lane-consecutive (conflict-free).
__global__ __launch_bounds__(256) void stageB1(
    const float* __restrict__ x1,
    const float* __restrict__ n2w, const float* __restrict__ n2b,
    const bf16_t* __restrict__ wpw,
    bf16_t* __restrict__ y0)   // (B, C, H*W) planar bf16
{
  extern __shared__ char smem[];
  float*  s_x  = (float*)smem;               // [64][256] f32 = 64KB (ch-major)
  bf16_t* s_xn = (bf16_t*)(s_x + CC * 256);  // [256][B1_LDB] bf16

  const int tid = threadIdx.x, wv = waveid();
  const int blk = blockIdx.x;
  const int b = blk / (HWP / 256);
  const int pix0 = (blk % (HWP / 256)) * 256;
  const float* xb = x1 + (size_t)b * CC * HWP;

#ifdef HAVE_ASYNC_LDS
#pragma unroll
  for (int i = 0; i < 16; ++i) {
    int g = i * 256 + tid;                    // 16B chunk id, 0..4095
    int ch = g >> 6, px4 = (g & 63) << 2;
    async_ld_b128(xb + (size_t)ch * HWP + pix0 + px4, s_x + g * 4);
  }
  async_wait0();
#else
#pragma unroll 4
  for (int i = 0; i < 64; ++i) {
    int f = i * 256 + tid;                    // flat f32 index (ch-major)
    s_x[f] = xb[(size_t)(f >> 8) * HWP + pix0 + (f & 255)];
  }
#endif
  __syncthreads();

  // LN per pixel (thread = pixel; column reads, lane-consecutive addresses)
  {
    int px = tid;
    float mu = 0.f;
    for (int ch = 0; ch < CC; ++ch) mu += s_x[ch * 256 + px];
    mu *= (1.f / CC);
    float var = 0.f;
    for (int ch = 0; ch < CC; ++ch) { float d = s_x[ch * 256 + px] - mu; var += d * d; }
    var *= (1.f / CC);
    float inv = rsqrtf(var + 1e-5f);
    for (int ch = 0; ch < CC; ++ch)
      s_xn[px * B1_LDB + ch] =
          (bf16_t)((s_x[ch * 256 + px] - mu) * inv * n2w[ch] + n2b[ch]);
  }
  __syncthreads();

  // (256x64) @ (64x64): 16 row-tiles x 4 col-tiles
  bf16_t* yb = y0 + (size_t)b * CC * HWP;
  for (int t = wv; t < 64; t += 8) {
    int rt = t >> 2, ct = t & 3;
    v8f acc = wmma_tile(s_xn, B1_LDB, rt * 16, wpw, CC, ct * 16, CC);
    int l = laneid();
    int ch = ct * 16 + (l & 15);
    int r0 = rt * 16 + ((l >> 4) << 3);
    bf16_t* dst = yb + (size_t)ch * HWP + pix0 + r0;
#pragma unroll
    for (int i = 0; i < 8; ++i) dst[i] = (bf16_t)acc[i];
  }
}

// ---------- Stage B2 : depthwise 3x3 SAME + bias ----------
__global__ __launch_bounds__(256) void stageB2(
    const bf16_t* __restrict__ y0, const float* __restrict__ dw,
    const float* __restrict__ dwb,
    bf16_t* __restrict__ y1)   // pixel-major (B*H*W, C) bf16
{
  int e = blockIdx.x * 256 + threadIdx.x;        // exactly B*C*H*W elements
  int w = e % WW;
  int t = e / WW;
  int h = t % HH;
  int t2 = t / HH;
  int ch = t2 % CC;
  int b  = t2 / CC;
  const bf16_t* plane = y0 + (size_t)(b * CC + ch) * HWP;
  const float* k = dw + ch * 9;
  float acc = dwb[ch];
#pragma unroll
  for (int ky = -1; ky <= 1; ++ky) {
    int hh = h + ky;
    if (hh < 0 || hh >= HH) continue;
#pragma unroll
    for (int kx = -1; kx <= 1; ++kx) {
      int wp = w + kx;
      if (wp < 0 || wp >= WW) continue;
      acc += (float)plane[hh * WW + wp] * k[(ky + 1) * 3 + (kx + 1)];
    }
  }
  y1[((size_t)b * HWP + (size_t)h * WW + w) * CC + ch] = (bf16_t)acc;
}

// ---------- Stage B34 : fc1 + exact GELU + fc2 + residual ----------
__global__ __launch_bounds__(256) void stageB34(
    const bf16_t* __restrict__ y1,
    const bf16_t* __restrict__ wfc1, const float* __restrict__ bfc1,
    const bf16_t* __restrict__ wfc2, const float* __restrict__ bfc2,
    float* __restrict__ xio)   // d_out: read x1, write final (RMW per element)
{
  extern __shared__ char smem[];
  bf16_t* s_a  = (bf16_t*)smem;        // [128][64] bf16 (unpadded: linear async copy)
  bf16_t* s_y2 = s_a + 128 * CC;       // [128][B3_LDY] bf16

  const int tid = threadIdx.x, wv = waveid();
  const int blk = blockIdx.x;
  const int b = blk / (HWP / 128);
  const int pix0 = (blk % (HWP / 128)) * 128;

  const bf16_t* src = y1 + ((size_t)b * HWP + pix0) * CC;   // 8192 contiguous bf16

#ifdef HAVE_ASYNC_LDS
#pragma unroll
  for (int i = 0; i < 4; ++i) {
    int g = (i * 256 + tid) * 8;       // bf16 elements, 16B per thread
    async_ld_b128(src + g, s_a + g);
  }
  async_wait0();
#else
#pragma unroll 4
  for (int i = 0; i < 32; ++i) s_a[i * 256 + tid] = src[i * 256 + tid];
#endif
  __syncthreads();

  // fc1: (128x64) @ (64x256) -> 8x16 tiles, exact GELU
  for (int t = wv; t < 128; t += 8) {
    int rt = t >> 4, ct = t & 15;
    v8f acc = wmma_tile(s_a, CC, rt * 16, wfc1, CC, ct * 16, CC);
    int l = laneid();
    int col = ct * 16 + (l & 15);
    int r0  = rt * 16 + ((l >> 4) << 3);
    float bv = bfc1[col];
#pragma unroll
    for (int i = 0; i < 8; ++i) {
      float v = acc[i] + bv;
      v = 0.5f * v * (1.f + erff(v * 0.70710678118654752f));
      s_y2[(r0 + i) * B3_LDY + col] = (bf16_t)v;
    }
  }
  __syncthreads();

  // fc2: (128x256) @ (256x64) -> 8x4 tiles, K=256; + bias + residual
  float* ob = xio + (size_t)b * CC * HWP;
  for (int t = wv; t < 32; t += 8) {
    int rt = t >> 2, ct = t & 3;
    v8f acc = wmma_tile(s_y2, B3_LDY, rt * 16, wfc2, 256, ct * 16, 256);
    int l = laneid();
    int ch = ct * 16 + (l & 15);
    int r0 = rt * 16 + ((l >> 4) << 3);
    float bv = bfc2[ch];
    float* dst = ob + (size_t)ch * HWP + pix0 + r0;
#pragma unroll
    for (int i = 0; i < 8; ++i) dst[i] = dst[i] + acc[i] + bv;
  }
}

// ---------- workspace layout (bytes) ----------
enum : size_t {
  OFF_WQ   = 0,          // 192*64  bf16
  OFF_WO   = 24576,
  OFF_WCP  = 32768,
  OFF_WPW1 = 40960,
  OFF_WTOK = 49152,
  OFF_WLPW = 57344,
  OFF_WFC1 = 65536,      // 256*64 bf16
  OFF_WFC2 = 98304,      // 64*256 bf16
  OFF_Y0   = 131072,     // B*C*HW bf16
  OFF_Y1   = OFF_Y0 + (size_t)BB * CC * HWP * 2,
};

// dynamic LDS sizes (bytes)
#define SMEM_A  (((SS * A_LDX + SS * A_LDQ + 3 * SS * A_LDR) * 4) + (3 * SS * A_LDB * 2))
#define SMEM_B1 ((CC * 256 * 4) + (256 * B1_LDB * 2))
#define SMEM_B3 ((128 * CC * 2) + (128 * B3_LDY * 2))

extern "C" void kernel_launch(void* const* d_in, const int* in_sizes, int n_in,
                              void* d_out, int out_size, void* d_ws, size_t ws_size,
                              hipStream_t stream) {
  (void)in_sizes; (void)n_in; (void)out_size; (void)ws_size;
  const float* x       = (const float*)d_in[0];
  const float* n1w     = (const float*)d_in[1];
  const float* n1b     = (const float*)d_in[2];
  const float* qkv_w   = (const float*)d_in[3];
  const float* qkv_b   = (const float*)d_in[4];
  const float* posb    = (const float*)d_in[5];
  const float* out_w   = (const float*)d_in[6];
  const float* out_b   = (const float*)d_in[7];
  const float* convp_w = (const float*)d_in[8];
  const float* convp_b = (const float*)d_in[9];
  const float* pw1_w   = (const float*)d_in[10];
  const float* pw1_b   = (const float*)d_in[11];
  const float* dw1_w   = (const float*)d_in[12];
  const float* dw1_b   = (const float*)d_in[13];
  const float* tok_w   = (const float*)d_in[14];
  const float* tok_b   = (const float*)d_in[15];
  const float* n2w     = (const float*)d_in[16];
  const float* n2b     = (const float*)d_in[17];
  const float* leff_pw = (const float*)d_in[18];
  const float* leff_dw = (const float*)d_in[19];
  const float* leff_db = (const float*)d_in[20];
  const float* fc1_w   = (const float*)d_in[21];
  const float* fc1_b   = (const float*)d_in[22];
  const float* fc2_w   = (const float*)d_in[23];
  const float* fc2_b   = (const float*)d_in[24];

  char* ws = (char*)d_ws;
  bf16_t* wq  = (bf16_t*)(ws + OFF_WQ);
  bf16_t* wo  = (bf16_t*)(ws + OFF_WO);
  bf16_t* wcp = (bf16_t*)(ws + OFF_WCP);
  bf16_t* wp1 = (bf16_t*)(ws + OFF_WPW1);
  bf16_t* wtk = (bf16_t*)(ws + OFF_WTOK);
  bf16_t* wlp = (bf16_t*)(ws + OFF_WLPW);
  bf16_t* wf1 = (bf16_t*)(ws + OFF_WFC1);
  bf16_t* wf2 = (bf16_t*)(ws + OFF_WFC2);
  bf16_t* y0  = (bf16_t*)(ws + OFF_Y0);
  bf16_t* y1  = (bf16_t*)(ws + OFF_Y1);
  float*  x1  = (float*)d_out;   // stageA writes x1 into d_out; B34 RMWs it into final output

  // weight conversions (tiny; resident in L2 afterwards)
  cvt_bf16<<<48, 256, 0, stream>>>(qkv_w, wq, 192 * CC);
  cvt_bf16<<<16, 256, 0, stream>>>(out_w, wo, CC * CC);
  cvt_bf16<<<16, 256, 0, stream>>>(convp_w, wcp, CC * CC);
  cvt_bf16<<<16, 256, 0, stream>>>(pw1_w, wp1, CC * CC);
  cvt_bf16<<<16, 256, 0, stream>>>(tok_w, wtk, CC * CC);
  cvt_bf16<<<16, 256, 0, stream>>>(leff_pw, wlp, CC * CC);
  cvt_bf16<<<64, 256, 0, stream>>>(fc1_w, wf1, 4 * CC * CC);
  cvt_bf16<<<64, 256, 0, stream>>>(fc2_w, wf2, 4 * CC * CC);

  // Stage A: one block per 8x8 window (2 * 48 * 48 = 4608)
  stageA<<<BB * NWX * NWY, 256, SMEM_A, stream>>>(
      x, n1w, n1b, wq, qkv_b, posb, wo, out_b, wcp, convp_b,
      wp1, pw1_b, dw1_w, dw1_b, wtk, tok_b, x1);

  // Stage B1: LN2 + 1x1 conv (2 * 576 blocks of 256 pixels)
  stageB1<<<BB * (HWP / 256), 256, SMEM_B1, stream>>>(x1, n2w, n2b, wlp, y0);

  // Stage B2: depthwise 3x3 (elementwise over B*C*H*W)
  stageB2<<<(BB * CC * HWP) / 256, 256, 0, stream>>>(y0, leff_dw, leff_db, y1);

  // Stage B34: fc1+GELU+fc2+residual (2 * 1152 blocks of 128 pixels)
  stageB34<<<BB * (HWP / 128), 256, SMEM_B3, stream>>>(y1, wf1, fc1_b, wf2, fc2_b, x1);
}
